// MultiHeadAttention_91190745629033
// MI455X (gfx1250) — matrix operand order repaired and checked by reading, not compiled, not run on hardware
//
#include <hip/hip_runtime.h>
#include <hip/hip_bf16.h>

// ---------------------------------------------------------------------------
// MultiHeadAttention forward for MI455X (gfx1250, wave32, WMMA).
// B=2, S=2048, D=1024, H=16, Dh=64.
// d_out = [ out (B*S*D f32) | attn (B*H*S*S f32) ]
// ---------------------------------------------------------------------------

typedef __attribute__((ext_vector_type(16))) __bf16        v16bf;
typedef __attribute__((ext_vector_type(8)))  float         v8f;
typedef unsigned short                                     u16;
typedef __attribute__((ext_vector_type(8)))  unsigned short u16x8;

union FragU {
    v16bf v;
    u16   u[16];
};

#define BB   2
#define SS   2048
#define DD   1024
#define HH   16
#define DHH  64
#define NEGV (-1e9f)

__device__ __forceinline__ u16 f2bf(float x) {
    unsigned u = __float_as_uint(x);
    u += 0x7FFFu + ((u >> 16) & 1u);   // round-to-nearest-even
    return (u16)(u >> 16);
}

// ---------------------------------------------------------------------------
// f32 -> bf16 elementwise convert
// ---------------------------------------------------------------------------
__global__ void cvt_bf16_kernel(const float* __restrict__ src,
                                u16* __restrict__ dst, int n) {
    int i = blockIdx.x * blockDim.x + threadIdx.x;
    if (i < n) dst[i] = f2bf(src[i]);
}

// f32 [K][N] -> bf16 transposed [N][K]
__global__ void cvt_bf16_T_kernel(const float* __restrict__ src,
                                  u16* __restrict__ dst, int K, int N) {
    int i = blockIdx.x * blockDim.x + threadIdx.x;
    if (i < K * N) {
        int k = i / N, n = i % N;
        dst[n * K + k] = f2bf(src[i]);
    }
}

// Vp bf16 [B*S][D] -> Vt bf16 [(b*H+h)*64+dh][S]
__global__ void transpose_v_kernel(const u16* __restrict__ Vp,
                                   u16* __restrict__ Vt) {
    int i = blockIdx.x * blockDim.x + threadIdx.x;
    if (i < BB * SS * DD) {
        int d = i % DD;
        int s = (i / DD) % SS;
        int b = i / (DD * SS);
        int h = d / DHH, dh = d % DHH;
        Vt[((size_t)((b * HH + h) * DHH + dh)) * SS + s] = Vp[i];
    }
}

// ---------------------------------------------------------------------------
// GEMM: C[M][N] = A[M][K] (bf16 row-major) @ Bt[N][K] (bf16, B pre-transposed)
// One 16x64 output strip per wave (4 accumulators share one A fragment);
// 8 waves per block cover a 16x512 strip.
// ---------------------------------------------------------------------------
__global__ void gemm_bf16_kernel(const u16* __restrict__ A,
                                 const u16* __restrict__ Bt,
                                 float* __restrict__ Cf,
                                 u16*   __restrict__ Cbf,
                                 int M, int N, int K) {
    const int wave = threadIdx.x >> 5;
    const int lane = threadIdx.x & 31;
    const int half = lane >> 4;
    const int r16  = lane & 15;

    const int m0 = blockIdx.x * 16;
    const int n0 = (blockIdx.y * 8 + wave) * 64;
    if (m0 >= M || n0 >= N) return;

    const u16* aRow = A + (size_t)(m0 + r16) * K + half * 16;
    const u16* bRow0 = Bt + (size_t)(n0 + 0  + r16) * K + half * 16;
    const u16* bRow1 = Bt + (size_t)(n0 + 16 + r16) * K + half * 16;
    const u16* bRow2 = Bt + (size_t)(n0 + 32 + r16) * K + half * 16;
    const u16* bRow3 = Bt + (size_t)(n0 + 48 + r16) * K + half * 16;

    const v8f vzero = {};
    v8f acc[4] = {vzero, vzero, vzero, vzero};

    for (int kk = 0; kk < K; kk += 32) {
        __builtin_prefetch(aRow + kk + 512, 0, 1);
        __builtin_prefetch(bRow0 + kk + 512, 0, 1);
        __builtin_prefetch(bRow2 + kk + 512, 0, 1);
        FragU a;
        *(u16x8*)&a.u[0] = *(const u16x8*)(aRow + kk);        // K kk+16h..+7
        *(u16x8*)&a.u[8] = *(const u16x8*)(aRow + kk + 8);    // K kk+16h+8..
        FragU b0, b1, b2, b3;
        *(u16x8*)&b0.u[0] = *(const u16x8*)(bRow0 + kk);
        *(u16x8*)&b0.u[8] = *(const u16x8*)(bRow0 + kk + 8);
        *(u16x8*)&b1.u[0] = *(const u16x8*)(bRow1 + kk);
        *(u16x8*)&b1.u[8] = *(const u16x8*)(bRow1 + kk + 8);
        *(u16x8*)&b2.u[0] = *(const u16x8*)(bRow2 + kk);
        *(u16x8*)&b2.u[8] = *(const u16x8*)(bRow2 + kk + 8);
        *(u16x8*)&b3.u[0] = *(const u16x8*)(bRow3 + kk);
        *(u16x8*)&b3.u[8] = *(const u16x8*)(bRow3 + kk + 8);
        acc[0] = __builtin_amdgcn_wmma_f32_16x16x32_bf16(
                     false, a.v, false, b0.v, (short)0, acc[0], false, false);
        acc[1] = __builtin_amdgcn_wmma_f32_16x16x32_bf16(
                     false, a.v, false, b1.v, (short)0, acc[1], false, false);
        acc[2] = __builtin_amdgcn_wmma_f32_16x16x32_bf16(
                     false, a.v, false, b2.v, (short)0, acc[2], false, false);
        acc[3] = __builtin_amdgcn_wmma_f32_16x16x32_bf16(
                     false, a.v, false, b3.v, (short)0, acc[3], false, false);
    }

#pragma unroll
    for (int t = 0; t < 4; ++t) {
        const int n = n0 + t * 16 + r16;
#pragma unroll
        for (int r = 0; r < 8; ++r) {
            const int m = m0 + r + half * 8;
            if (Cf)  Cf[(size_t)m * N + n]  = acc[t][r];
            if (Cbf) Cbf[(size_t)m * N + n] = f2bf(acc[t][r]);
        }
    }
}

// ---------------------------------------------------------------------------
// Flash-style attention: one wave per (b, h, 16-row q-tile).
// K-panels are staged global->LDS with async loads (ASYNCcnt engine), score
// B-fragments then come from ds_load_b128. Streams raw scaled+masked scores
// to `attn` (normalized later), keeps online row max/sum, accumulates context
// via P@V WMMA through an LDS relayout of P.
// ---------------------------------------------------------------------------
__global__ void attention_kernel(const u16* __restrict__ Qp,
                                 const u16* __restrict__ Kp,
                                 const u16* __restrict__ Vt,
                                 const unsigned char* __restrict__ mask,
                                 float* __restrict__ attn,
                                 u16*   __restrict__ ctx,
                                 float* __restrict__ stats) {
    __shared__ alignas(32) u16 Kt[32 * 64];   // K panel: 32 rows x Dh=64, 4 KiB
    __shared__ alignas(32) u16 Pt[16 * 32];   // P relayout tile, 1 KiB

    const int lane = threadIdx.x & 31;
    const int half = lane >> 4;
    const int r16  = lane & 15;
    const int q0 = blockIdx.x * 16;
    const int h  = blockIdx.y;
    const int b  = blockIdx.z;

    // Q A-fragments for the two K-chunks of Dh=64
    const u16* qRow = Qp + ((size_t)(b * SS) + q0 + r16) * DD + h * DHH + half * 16;
    FragU qa0, qa1;
    *(u16x8*)&qa0.u[0] = *(const u16x8*)(qRow + 0);
    *(u16x8*)&qa0.u[8] = *(const u16x8*)(qRow + 8);
    *(u16x8*)&qa1.u[0] = *(const u16x8*)(qRow + 32);
    *(u16x8*)&qa1.u[8] = *(const u16x8*)(qRow + 40);

    float m[8], l[8];
    v8f o[4];
    const v8f vzero = {};
#pragma unroll
    for (int r = 0; r < 8; ++r) { m[r] = -3.0e38f; l[r] = 0.f; }
#pragma unroll
    for (int t = 0; t < 4; ++t) o[t] = vzero;

    const u16* kBase = Kp + (size_t)(b * SS) * DD + h * DHH;
    const u16* vBase = Vt + (size_t)((b * HH + h) * DHH) * SS;
    float* attnBase  = attn + ((size_t)((b * HH + h) * SS) + q0) * SS;
    const unsigned char* mBase = mask + (size_t)b * SS * SS + (size_t)q0 * SS;

    // LDS byte offset of this lane's K-panel row (generic LDS ptr low 32 bits)
    const unsigned ktRow = (unsigned)(size_t)(void*)&Kt[lane * DHH];

    for (int j = 0; j < SS; j += 32) {
        // --- async-stage K panel rows j..j+31 into LDS ---------------------
        // INST_OFFSET applies to both the global and LDS addresses, so one
        // immediate walks both pointers across the 128-byte row.
        {
            const u16* gk = kBase + (size_t)(j + lane) * DD;   // one row/lane
            asm volatile(
                "global_load_async_to_lds_b128 %0, %1, off\n\t"
                "global_load_async_to_lds_b128 %0, %1, off offset:16\n\t"
                "global_load_async_to_lds_b128 %0, %1, off offset:32\n\t"
                "global_load_async_to_lds_b128 %0, %1, off offset:48\n\t"
                "global_load_async_to_lds_b128 %0, %1, off offset:64\n\t"
                "global_load_async_to_lds_b128 %0, %1, off offset:80\n\t"
                "global_load_async_to_lds_b128 %0, %1, off offset:96\n\t"
                "global_load_async_to_lds_b128 %0, %1, off offset:112"
                :: "v"(ktRow), "v"(gk) : "memory");
            asm volatile("s_wait_asynccnt 0x0" ::: "memory");
        }
        __syncthreads();

        // --- scores: two 16x16 tiles (cols j..j+15, j+16..j+31) ------------
        v8f s0 = vzero, s1 = vzero;
        const u16* k0 = &Kt[r16 * DHH + half * 16];         // LDS rows 0..15
        const u16* k1 = k0 + 16 * DHH;                      // LDS rows 16..31
        FragU b00, b01, b10, b11;
        *(u16x8*)&b00.u[0] = *(const u16x8*)(k0);
        *(u16x8*)&b00.u[8] = *(const u16x8*)(k0 + 8);
        *(u16x8*)&b01.u[0] = *(const u16x8*)(k0 + 32);
        *(u16x8*)&b01.u[8] = *(const u16x8*)(k0 + 40);
        *(u16x8*)&b10.u[0] = *(const u16x8*)(k1);
        *(u16x8*)&b10.u[8] = *(const u16x8*)(k1 + 8);
        *(u16x8*)&b11.u[0] = *(const u16x8*)(k1 + 32);
        *(u16x8*)&b11.u[8] = *(const u16x8*)(k1 + 40);

        s0 = __builtin_amdgcn_wmma_f32_16x16x32_bf16(false, qa0.v, false, b00.v,
                                                     (short)0, s0, false, false);
        s0 = __builtin_amdgcn_wmma_f32_16x16x32_bf16(false, qa1.v, false, b01.v,
                                                     (short)0, s0, false, false);
        s1 = __builtin_amdgcn_wmma_f32_16x16x32_bf16(false, qa0.v, false, b10.v,
                                                     (short)0, s1, false, false);
        s1 = __builtin_amdgcn_wmma_f32_16x16x32_bf16(false, qa1.v, false, b11.v,
                                                     (short)0, s1, false, false);

        // --- scale, mask, stream raw scores, online softmax ---------------
#pragma unroll
        for (int r = 0; r < 8; ++r) {
            const int mrow = r + half * 8;
            const int c0 = j + r16;
            const int c1 = j + 16 + r16;
            float v0 = s0[r] * 0.125f;   // 1/sqrt(64)
            float v1 = s1[r] * 0.125f;
            if (mBase[(size_t)mrow * SS + c0]) v0 = NEGV;
            if (mBase[(size_t)mrow * SS + c1]) v1 = NEGV;
            attnBase[(size_t)mrow * SS + c0] = v0;
            attnBase[(size_t)mrow * SS + c1] = v1;

            float t = fmaxf(v0, v1);
#pragma unroll
            for (int off = 1; off < 16; off <<= 1)
                t = fmaxf(t, __shfl_xor(t, off));
            const float mn   = fmaxf(m[r], t);
            const float corr = __expf(m[r] - mn);
            const float p0   = __expf(v0 - mn);
            const float p1   = __expf(v1 - mn);
            float rs = p0 + p1;
#pragma unroll
            for (int off = 1; off < 16; off <<= 1)
                rs += __shfl_xor(rs, off);
            l[r] = l[r] * corr + rs;
            m[r] = mn;
            o[0][r] *= corr; o[1][r] *= corr; o[2][r] *= corr; o[3][r] *= corr;

            Pt[mrow * 32 + r16]      = f2bf(p0);
            Pt[mrow * 32 + 16 + r16] = f2bf(p1);
        }
        __syncthreads();   // LDS store -> load visibility

        // --- P (16x32) @ V-tile (32x64): relayout P via LDS ----------------
        FragU pa;
        const u16* pRow = Pt + r16 * 32 + half * 16;
        *(u16x8*)&pa.u[0] = *(const u16x8*)(pRow);
        *(u16x8*)&pa.u[8] = *(const u16x8*)(pRow + 8);

#pragma unroll
        for (int t = 0; t < 4; ++t) {
            const u16* vp = vBase + (size_t)(t * 16 + r16) * SS + j + half * 16;
            FragU vb;
            *(u16x8*)&vb.u[0] = *(const u16x8*)(vp);
            *(u16x8*)&vb.u[8] = *(const u16x8*)(vp + 8);
            o[t] = __builtin_amdgcn_wmma_f32_16x16x32_bf16(
                       false, pa.v, false, vb.v, (short)0, o[t], false, false);
        }
        __syncthreads();   // protect Pt/Kt before next iteration overwrites
    }

    // --- epilogue: normalize context, emit per-row softmax stats -----------
#pragma unroll
    for (int r = 0; r < 8; ++r) {
        const int mrow = r + half * 8;
        const float inv = 1.0f / l[r];
        if (r16 == 0) {
            const size_t rowg = (size_t)((b * HH + h) * SS) + q0 + mrow;
            stats[2 * rowg]     = m[r];
            stats[2 * rowg + 1] = l[r];
        }
#pragma unroll
        for (int t = 0; t < 4; ++t) {
            ctx[((size_t)(b * SS) + q0 + mrow) * DD + h * DHH + t * 16 + r16] =
                f2bf(o[t][r] * inv);
        }
    }
}

// ---------------------------------------------------------------------------
// attn[i] = exp(raw - m_row) / l_row
// ---------------------------------------------------------------------------
__global__ void normalize_attn_kernel(float* __restrict__ attn,
                                      const float* __restrict__ stats,
                                      long long n) {
    long long i = (long long)blockIdx.x * blockDim.x + threadIdx.x;
    const long long stride = (long long)gridDim.x * blockDim.x;
    for (; i < n; i += stride) {
        const long long row = i >> 11;      // / 2048
        const float mm = stats[2 * row];
        const float ll = stats[2 * row + 1];
        attn[i] = __expf(attn[i] - mm) / ll;
    }
}

// ---------------------------------------------------------------------------
// out = LayerNorm(fc + input_Q), gamma=1 beta=0, eps=1e-5. One row per block.
// ---------------------------------------------------------------------------
__global__ void layernorm_kernel(const float* __restrict__ fc,
                                 const float* __restrict__ inQ,
                                 float* __restrict__ out) {
    const int row = blockIdx.x;
    const float* f = fc  + (size_t)row * DD;
    const float* q = inQ + (size_t)row * DD;
    float* o       = out + (size_t)row * DD;

    float s = 0.f, ss = 0.f;
    for (int i = threadIdx.x; i < DD; i += blockDim.x) {
        const float y = f[i] + q[i];
        s += y; ss += y * y;
    }
#pragma unroll
    for (int off = 16; off > 0; off >>= 1) {
        s  += __shfl_xor(s, off);
        ss += __shfl_xor(ss, off);
    }
    __shared__ float rs[8], rss[8];
    __shared__ float smean, srstd;
    const int w = threadIdx.x >> 5;
    if ((threadIdx.x & 31) == 0) { rs[w] = s; rss[w] = ss; }
    __syncthreads();
    if (threadIdx.x == 0) {
        float S = 0.f, SQ = 0.f;
        for (int i = 0; i < 8; ++i) { S += rs[i]; SQ += rss[i]; }
        const float mu  = S * (1.0f / DD);
        const float var = SQ * (1.0f / DD) - mu * mu;
        smean = mu;
        srstd = rsqrtf(var + 1e-5f);
    }
    __syncthreads();
    const float mu = smean, rstd = srstd;
    for (int i = threadIdx.x; i < DD; i += blockDim.x)
        o[i] = (f[i] + q[i] - mu) * rstd;
}

// ---------------------------------------------------------------------------
// Host orchestration
// ---------------------------------------------------------------------------
extern "C" void kernel_launch(void* const* d_in, const int* in_sizes, int n_in,
                              void* d_out, int out_size, void* d_ws, size_t ws_size,
                              hipStream_t stream) {
    const float* inQ  = (const float*)d_in[0];
    const float* inK  = (const float*)d_in[1];
    const float* inV  = (const float*)d_in[2];
    const unsigned char* mask = (const unsigned char*)d_in[3];
    const float* WQ   = (const float*)d_in[4];
    const float* WK   = (const float*)d_in[5];
    const float* WV   = (const float*)d_in[6];
    const float* Wfc  = (const float*)d_in[7];

    float* out  = (float*)d_out;                       // [B*S*D]
    float* attn = out + (size_t)BB * SS * DD;          // [B*H*S*S]

    // Workspace layout (bytes). Peak usage ~64 MB with aggressive reuse.
    char* w = (char*)d_ws;
    const size_t ACT  = (size_t)BB * SS * DD * 2;      // 8 MiB bf16 activation
    const size_t WGT  = (size_t)DD * DD * 2;           // 2 MiB bf16 weight
    u16* Qb  = (u16*)(w);                              // bf16 input_Q
    u16* Kb  = (u16*)(w + ACT);
    u16* Vb  = (u16*)(w + 2 * ACT);
    u16* WQt = (u16*)(w + 3 * ACT);
    u16* WKt = (u16*)(w + 3 * ACT + WGT);
    u16* WVt = (u16*)(w + 3 * ACT + 2 * WGT);
    u16* WFt = (u16*)(w + 3 * ACT + 3 * WGT);
    u16* Qp  = (u16*)(w + 3 * ACT + 4 * WGT);          // projected Q bf16
    u16* Kp  = (u16*)(w + 4 * ACT + 4 * WGT);
    u16* Vp  = (u16*)(w + 5 * ACT + 4 * WGT);
    u16* Vt  = (u16*)(w + 6 * ACT + 4 * WGT);          // V transposed per head
    // Reuse dead regions after projections:
    u16*   Ctx   = Qb;                                 // context bf16 (Qb dead)
    float* FCout = (float*)(w + ACT);                  // fc f32 (Kb+Vb dead)
    float* Stats = (float*)(w + 3 * ACT);              // m,l pairs (WQt dead)

    const int nAct = BB * SS * DD;                     // 4,194,304
    const int nW   = DD * DD;                          // 1,048,576
    const dim3 cvtB(256);

    // 1) bf16 conversions + weight transposes
    cvt_bf16_kernel<<<(nAct + 255) / 256, cvtB, 0, stream>>>(inQ, Qb, nAct);
    cvt_bf16_kernel<<<(nAct + 255) / 256, cvtB, 0, stream>>>(inK, Kb, nAct);
    cvt_bf16_kernel<<<(nAct + 255) / 256, cvtB, 0, stream>>>(inV, Vb, nAct);
    cvt_bf16_T_kernel<<<(nW + 255) / 256, cvtB, 0, stream>>>(WQ,  WQt, DD, DD);
    cvt_bf16_T_kernel<<<(nW + 255) / 256, cvtB, 0, stream>>>(WK,  WKt, DD, DD);
    cvt_bf16_T_kernel<<<(nW + 255) / 256, cvtB, 0, stream>>>(WV,  WVt, DD, DD);
    cvt_bf16_T_kernel<<<(nW + 255) / 256, cvtB, 0, stream>>>(Wfc, WFt, DD, DD);

    // 2) Q/K/V projections: [4096x1024] = [4096x1024] @ [1024x1024]
    const int M = BB * SS;
    dim3 gGemm(M / 16, DD / 512);       // 16x64 strip per wave, 8 waves/block
    dim3 bGemm(256);
    gemm_bf16_kernel<<<gGemm, bGemm, 0, stream>>>(Qb, WQt, nullptr, Qp, M, DD, DD);
    gemm_bf16_kernel<<<gGemm, bGemm, 0, stream>>>(Kb, WKt, nullptr, Kp, M, DD, DD);
    gemm_bf16_kernel<<<gGemm, bGemm, 0, stream>>>(Vb, WVt, nullptr, Vp, M, DD, DD);

    // 3) V per-head transpose for contiguous B-fragment loads in P@V
    transpose_v_kernel<<<(nAct + 255) / 256, cvtB, 0, stream>>>(Vp, Vt);

    // 4) attention (raw scores -> attn, bf16 context, softmax stats)
    dim3 gAttn(SS / 16, HH, BB);
    attention_kernel<<<gAttn, dim3(32), 0, stream>>>(Qp, Kp, Vt, mask,
                                                     attn, Ctx, Stats);

    // 5) softmax normalization of streamed scores
    const long long nAttn = (long long)BB * HH * SS * SS;
    normalize_attn_kernel<<<4096, 256, 0, stream>>>(attn, Stats, nAttn);

    // 6) output projection
    gemm_bf16_kernel<<<gGemm, bGemm, 0, stream>>>(Ctx, WFt, FCout, nullptr, M, DD, DD);

    // 7) residual + LayerNorm
    layernorm_kernel<<<M, 256, 0, stream>>>(FCout, inQ, out);
}
